// NeuMF_46505905881486
// MI455X (gfx1250) — matrix operand order repaired and checked by reading, not compiled, not run on hardware
//
#include <hip/hip_runtime.h>
#include <hip/hip_bf16.h>

// ---------------------------------------------------------------------------
// NeuMF fused forward for gfx1250 (MI455X, wave32, WMMA).
//
// Per block: 64 batch rows. Gather MLP embeddings -> LDS (f16), run the
// 512->256->128->64 ReLU tower with v_wmma_f32_16x16x32_f16 (f32 accum),
// then fuse GMF dot + prediction head.  Weights (688 KB total) live in L2;
// each block stages 64-column transposed f16 chunks into LDS.
// ---------------------------------------------------------------------------

typedef __attribute__((ext_vector_type(16))) _Float16 v16h;
typedef __attribute__((ext_vector_type(8)))  _Float16 v8h;
typedef __attribute__((ext_vector_type(4)))  _Float16 v4h;
typedef __attribute__((ext_vector_type(8)))  float    v8f;
typedef __attribute__((ext_vector_type(4)))  float    v4f;

#define BATCH_N    65536
#define ROWS       64          // batch rows per workgroup
#define THREADS    256         // 8 wave32s
#define ACT_STRIDE 520         // f16 elems per activation row (512 + 8 pad -> conflict-free b128)
#define WT_STRIDE  520         // f16 elems per staged weight column (Kmax 512 + 8 pad)
#define NCHUNK     64          // weight columns staged per LDS chunk

// A fragment (16x32 f16, ISA layout): lane l (g = l>>4, m = l&15) holds row
// M=m, K = {kb+8g .. kb+8g+7} ++ {kb+16+8g .. kb+16+8g+7}  -> two b128 loads.
__device__ __forceinline__ v16h load_a_frag(const _Float16* act, int row, int kb, int g) {
    const _Float16* p = act + row * ACT_STRIDE + kb + g * 8;
    v8h lo = *(const v8h*)p;
    v8h hi = *(const v8h*)(p + 16);
    return __builtin_shufflevector(lo, hi, 0, 1, 2, 3, 4, 5, 6, 7, 8, 9, 10, 11, 12, 13, 14, 15);
}

// B fragment (32x16 f16, ISA layout): lane l (g = l>>4, n = l&15) holds col N=n,
// K = {kb+8g .. kb+8g+7} ++ {kb+16+8g .. kb+16+8g+7} (column-major staged weights) -> two b128 loads.
__device__ __forceinline__ v16h load_b_frag(const _Float16* wt, int ncol, int kb, int g) {
    const _Float16* p = wt + ncol * WT_STRIDE + kb + g * 8;
    v8h lo = *(const v8h*)p;
    v8h hi = *(const v8h*)(p + 16);
    return __builtin_shufflevector(lo, hi, 0, 1, 2, 3, 4, 5, 6, 7, 8, 9, 10, 11, 12, 13, 14, 15);
}

// One Linear(K->N) + bias + ReLU layer on a 64-row tile, f16 in/out in LDS.
__device__ __forceinline__ void mlp_layer(const _Float16* __restrict__ in,
                                          _Float16* __restrict__ out,
                                          _Float16* __restrict__ wT,
                                          const float* __restrict__ W,
                                          const float* __restrict__ bias,
                                          int K, int N) {
    const int tid  = threadIdx.x;
    const int lane = tid & 31;
    const int wave = tid >> 5;
    const int g    = lane >> 4;
    const int ln   = lane & 15;

    for (int nc = 0; nc < N; nc += NCHUNK) {
        __syncthreads();   // previous chunk's WMMA reads of wT done
        // ---- stage 64 columns of W: global f32 row-major -> LDS f16 col-major
        const int stage_iters = K * (NCHUNK / 4);     // float4 granules
        for (int i = tid; i < stage_iters; i += THREADS) {
            int k  = i >> 4;                          // NCHUNK/4 == 16
            int n4 = (i & 15) << 2;
            v4f w = *(const v4f*)(W + (long)k * N + nc + n4);
            wT[(n4 + 0) * WT_STRIDE + k] = (_Float16)w.x;
            wT[(n4 + 1) * WT_STRIDE + k] = (_Float16)w.y;
            wT[(n4 + 2) * WT_STRIDE + k] = (_Float16)w.z;
            wT[(n4 + 3) * WT_STRIDE + k] = (_Float16)w.w;
        }
        // warm L2/WGP$ for the next chunk while we compute this one
        if (nc + NCHUNK < N) __builtin_prefetch(W + nc + NCHUNK + (long)tid * N, 0, 1);
        __syncthreads();

        // ---- 16 output tiles (4 row-tiles x 4 col-tiles of 16x16); 2 per wave
        for (int t = wave; t < 16; t += 8) {
            int rt = t >> 2;
            int ct = t & 3;
            v8f c = {};
            for (int kb = 0; kb < K; kb += 32) {
                v16h a = load_a_frag(in, rt * 16 + ln, kb, g);
                v16h b = load_b_frag(wT, ct * 16 + ln, kb, g);
                c = __builtin_amdgcn_wmma_f32_16x16x32_f16(
                        /*neg_a=*/false, a, /*neg_b=*/false, b,
                        /*c_mod=*/(short)0, c, /*reuse_a=*/false, /*reuse_b=*/false);
            }
            int ng   = nc + ct * 16 + ln;             // global output column
            float bv = bias[ng];
            // C layout: lanes 0-15 N=ln, vgpr r -> M=r ; lanes 16-31 -> M=r+8
#pragma unroll
            for (int r = 0; r < 8; ++r) {
                float x = c[r] + bv;
                x = x > 0.0f ? x : 0.0f;
                out[(rt * 16 + r + g * 8) * ACT_STRIDE + ng] = (_Float16)x;
            }
        }
    }
}

__global__ __launch_bounds__(THREADS)
void neumf_fused_kernel(const int* __restrict__ user, const int* __restrict__ item,
                        const float* __restrict__ mf_user, const float* __restrict__ mf_item,
                        const float* __restrict__ mlp_user, const float* __restrict__ mlp_item,
                        const float* __restrict__ W1, const float* __restrict__ W2,
                        const float* __restrict__ W3,
                        const float* __restrict__ b1, const float* __restrict__ b2,
                        const float* __restrict__ b3,
                        const float* __restrict__ pW, const float* __restrict__ pb,
                        float* __restrict__ out) {
    extern __shared__ __attribute__((aligned(16))) char smem_raw[];
    _Float16* buf0 = (_Float16*)smem_raw;               // [64][520] f16
    _Float16* buf1 = buf0 + ROWS * ACT_STRIDE;          // [64][520] f16
    _Float16* wT   = buf1 + ROWS * ACT_STRIDE;          // [64][520] f16 (weight chunk, col-major)
    int* uIdx = (int*)(wT + NCHUNK * WT_STRIDE);
    int* iIdx = uIdx + ROWS;

    const int tid  = threadIdx.x;
    const int lane = tid & 31;
    const int wave = tid >> 5;
    const int row0 = blockIdx.x * ROWS;

    // ---- indices for this row tile
    if (tid < ROWS) {
        uIdx[tid] = user[row0 + tid];
        iIdx[tid] = item[row0 + tid];
    }
    __syncthreads();

    // ---- gather MLP embeddings -> buf0 (f16): concat(user 256, item 256)
    // one wave per 8 rows; 32 lanes * float4 covers 128 floats per pass
#pragma unroll
    for (int rr = 0; rr < 8; ++rr) {
        int row = wave * 8 + rr;
        const float* pu = mlp_user + (long)uIdx[row] * 256;
        const float* pi = mlp_item + (long)iIdx[row] * 256;
#pragma unroll
        for (int part = 0; part < 2; ++part) {
            int col = part * 128 + lane * 4;
            v4f vu = *(const v4f*)(pu + col);
            v4f vi = *(const v4f*)(pi + col);
            v4h hu = {(_Float16)vu.x, (_Float16)vu.y, (_Float16)vu.z, (_Float16)vu.w};
            v4h hi = {(_Float16)vi.x, (_Float16)vi.y, (_Float16)vi.z, (_Float16)vi.w};
            *(v4h*)(buf0 + row * ACT_STRIDE + col)       = hu;
            *(v4h*)(buf0 + row * ACT_STRIDE + 256 + col) = hi;
        }
    }

    // ---- MLP tower (ping-pong buffers); each layer self-synchronizes
    mlp_layer(buf0, buf1, wT, W1, b1, 512, 256);
    mlp_layer(buf1, buf0, wT, W2, b2, 256, 128);
    mlp_layer(buf0, buf1, wT, W3, b3, 128, 64);
    __syncthreads();

    // ---- prediction head: out = mf_u . (mf_i * pW[0:64]) + h3 . pW[64:128] + pb
    if (tid < ROWS) {
        int row = tid;
        const float* pu = mf_user + (long)uIdx[row] * 64;
        const float* pi = mf_item + (long)iIdx[row] * 64;
        float acc = pb[0];
#pragma unroll 4
        for (int d = 0; d < 64; ++d) acc += pu[d] * pi[d] * pW[d];
        const _Float16* h3 = buf1 + row * ACT_STRIDE;
#pragma unroll 4
        for (int d = 0; d < 64; ++d) acc += (float)h3[d] * pW[64 + d];
        out[row0 + row] = acc;
    }
}

extern "C" void kernel_launch(void* const* d_in, const int* in_sizes, int n_in,
                              void* d_out, int out_size, void* d_ws, size_t ws_size,
                              hipStream_t stream) {
    (void)in_sizes; (void)n_in; (void)out_size; (void)d_ws; (void)ws_size;
    const int*   user     = (const int*)d_in[0];
    const int*   item     = (const int*)d_in[1];
    const float* mf_user  = (const float*)d_in[2];
    const float* mf_item  = (const float*)d_in[3];
    const float* mlp_user = (const float*)d_in[4];
    const float* mlp_item = (const float*)d_in[5];
    const float* W1 = (const float*)d_in[6];
    const float* W2 = (const float*)d_in[7];
    const float* W3 = (const float*)d_in[8];
    const float* b1 = (const float*)d_in[9];
    const float* b2 = (const float*)d_in[10];
    const float* b3 = (const float*)d_in[11];
    const float* pW = (const float*)d_in[12];
    const float* pb = (const float*)d_in[13];
    float* out = (float*)d_out;

    // LDS: 3 * 64*520 f16 buffers + 2*64 int indices  (~196 KB; gfx1250 WGP = 320 KB)
    size_t lds_bytes = (size_t)(3 * ROWS * ACT_STRIDE) * sizeof(_Float16) + 2 * ROWS * sizeof(int);
    dim3 grid(BATCH_N / ROWS);
    dim3 block(THREADS);
    neumf_fused_kernel<<<grid, block, lds_bytes, stream>>>(
        user, item, mf_user, mf_item, mlp_user, mlp_item,
        W1, W2, W3, b1, b2, b3, pW, pb, out);
}
